// SpecifyKeyAttLayer_21715354648956
// MI455X (gfx1250) — compile-verified, matrix operand order checked by reading, not run on hardware
//
#include <hip/hip_runtime.h>

#define B_DIM 128
#define S_DIM 1024
#define D_DIM 256
#define ATT   128
#define EPSF  1e-7f
#define WT_PAD_U4 33   // padded row stride in uint4 (32 data + 1 pad) -> bank-conflict-free

typedef __attribute__((ext_vector_type(16))) __bf16 v16bf;
typedef __attribute__((ext_vector_type(8)))  float  v8f;

union Frag {
  v16bf v;
  unsigned short h[16];
  uint4 q[2];
};

// Native-precision fp32 -> bf16 (RNE) via HW convert; backend picks v_cvt*bf16
__device__ __forceinline__ void cvt8(Frag& fr, int base, float4 a, float4 b) {
  fr.v[base + 0] = (__bf16)a.x;  fr.v[base + 1] = (__bf16)a.y;
  fr.v[base + 2] = (__bf16)a.z;  fr.v[base + 3] = (__bf16)a.w;
  fr.v[base + 4] = (__bf16)b.x;  fr.v[base + 5] = (__bf16)b.y;
  fr.v[base + 6] = (__bf16)b.z;  fr.v[base + 7] = (__bf16)b.w;
}

__device__ __forceinline__ float fast_tanh(float x) {
#if __has_builtin(__builtin_amdgcn_tanhf)
  return __builtin_amdgcn_tanhf(x);   // V_TANH_F32 (trans op, co-executes with WMMA)
#else
  return tanhf(x);
#endif
}

// exact RNE for the one-time W conversion (prep kernel, cold path)
__device__ __forceinline__ unsigned short f2bf_bits(float f) {
  unsigned int x = __float_as_uint(f);
  x += 0x7FFFu + ((x >> 16) & 1u);
  return (unsigned short)(x >> 16);
}

// ---------------------------------------------------------------------------
// Prep: zero accumulators, u = W_u @ query, W -> bf16 transposed Wt[att][d]
// ---------------------------------------------------------------------------
__global__ void prep_kernel(const float* __restrict__ query,
                            const float* __restrict__ W,
                            const float* __restrict__ Wu,
                            float* __restrict__ zero_region,   // num (32768) + den (128)
                            float* __restrict__ uvec,
                            unsigned short* __restrict__ Wt) {
  const int stride = gridDim.x * blockDim.x;
  for (int idx = blockIdx.x * blockDim.x + threadIdx.x; idx < 65536; idx += stride) {
    if (idx < 32896) zero_region[idx] = 0.0f;
    if (idx < ATT) {
      float s = 0.0f;
      #pragma unroll 8
      for (int q = 0; q < ATT; ++q) s += Wu[idx * ATT + q] * query[q];
      uvec[idx] = s;
    }
    if (idx < ATT * D_DIM) {
      const int a = idx >> 8;    // att row of Wt
      const int k = idx & 255;   // d column of Wt
      Wt[idx] = f2bf_bits(W[k * ATT + a]);   // W is (D, ATT) row-major
    }
  }
}

// ---------------------------------------------------------------------------
// Main: fused scores (bf16 WMMA) + exp*mask + partial den / num accumulation
// grid = 512 (batch b = bx>>2, s-chunk of 256 rows = (bx&3)*256), block = 128
// ---------------------------------------------------------------------------
__global__ __launch_bounds__(128)
void attn_main_kernel(const float* __restrict__ key,
                      const int* __restrict__ mask,
                      const float* __restrict__ bias,
                      const float* __restrict__ uvec,
                      const unsigned short* __restrict__ Wt,
                      float* __restrict__ num,
                      float* __restrict__ den) {
  __shared__ uint4 sWt[ATT * WT_PAD_U4];   // 67584 B, bf16 Wt[att][d], padded rows
  __shared__ float sU[ATT];
  __shared__ float sB[ATT];
  __shared__ float sw[256];
  __shared__ float rbuf[4];

  const int tid  = threadIdx.x;
  const int wave = tid >> 5;
  const int lane = tid & 31;
  const int half = lane >> 4;   // wave32: lane half selects K lo/hi in fragments
  const int l16  = lane & 15;

  const int b  = blockIdx.x >> 2;
  const int s0 = (blockIdx.x & 3) * 256;
  const float* keyb = key + (size_t)b * S_DIM * D_DIM;

  // Stage bf16 Wt into LDS (padded row stride)
  {
    const uint4* src = (const uint4*)Wt;
    #pragma unroll 1
    for (int i = tid; i < ATT * 32; i += 128) {
      const int row = i >> 5;
      const int col = i & 31;
      sWt[row * WT_PAD_U4 + col] = src[i];
    }
  }
  if (tid < ATT) { sU[tid] = uvec[tid]; sB[tid] = bias[tid]; }
  __syncthreads();

  const unsigned short* sWt16 = (const unsigned short*)sWt;

  // --------------------- Phase A: scores -> sw[256] -----------------------
  #pragma unroll 1            // keep one tile's working set live (VGPR pressure)
  for (int tile = wave; tile < 16; tile += 4) {
    const int srow = s0 + tile * 16;
    const int arow = srow + l16;                 // A layout: lane l16 = row M
    const float* rp = keyb + (size_t)arow * D_DIM;
    const int kb0 = half * 8;                    // upper half-wave: K+8 / K+24

    if (tile + 4 < 16)
      __builtin_prefetch((const void*)(keyb + (size_t)(arow + 64) * D_DIM + kb0), 0, 1);

    // A fragments: key rows as bf16, ISA 16-bit A 16x32 layout, all K=256
    Frag afr[8];
    #pragma unroll
    for (int c8 = 0; c8 < 8; ++c8) {
      const int kb = c8 * 32 + kb0;
      const float4 f0 = *(const float4*)(rp + kb);
      const float4 f1 = *(const float4*)(rp + kb + 4);
      const float4 f2 = *(const float4*)(rp + kb + 16);
      const float4 f3 = *(const float4*)(rp + kb + 20);
      cvt8(afr[c8], 0, f0, f1);
      cvt8(afr[c8], 8, f2, f3);
    }

    float p[8] = {0.f,0.f,0.f,0.f,0.f,0.f,0.f,0.f};

    #pragma unroll
    for (int nt = 0; nt < 8; ++nt) {            // 8 N-tiles cover ATT=128
      const int n = nt * 16 + l16;              // B layout: lane l16 = column N
      const unsigned short* wrow = sWt16 + n * (WT_PAD_U4 * 8);
      v8f acc = {};
      #pragma unroll
      for (int c8 = 0; c8 < 8; ++c8) {          // K = 256 in steps of 32
        const int kb = c8 * 32 + kb0;
        Frag bfr;
        bfr.q[0] = *(const uint4*)(wrow + kb);        // K=kb..kb+7
        bfr.q[1] = *(const uint4*)(wrow + kb + 16);   // K=kb+16..kb+23
        acc = __builtin_amdgcn_wmma_f32_16x16x32_bf16(
            false, afr[c8].v, false, bfr.v, (short)0, acc, false, false);
      }
      // C layout: VGPR r -> rows r / r+8, column = l16. tanh(+bias) dot u.
      const float un = sU[n];
      const float bn = sB[n];
      #pragma unroll
      for (int r = 0; r < 8; ++r) p[r] += un * fast_tanh(acc[r] + bn);
    }

    // Reduce over the 16 columns (lanes within each half-wave group)
    #pragma unroll
    for (int r = 0; r < 8; ++r) {
      p[r] += __shfl_xor(p[r], 1, 32);
      p[r] += __shfl_xor(p[r], 2, 32);
      p[r] += __shfl_xor(p[r], 4, 32);
      p[r] += __shfl_xor(p[r], 8, 32);
    }

    if (l16 == 0) {                             // lane0 -> rows 0..7, lane16 -> rows 8..15
      #pragma unroll
      for (int r = 0; r < 8; ++r) {
        const int rloc = tile * 16 + half * 8 + r;
        const float m = (float)mask[b * S_DIM + (s0 + rloc)];
        sw[rloc] = __expf(p[r]) * m;
      }
    }
  }
  __syncthreads();

  // --------------------- Phase B: den / num partials ----------------------
  float part = 0.0f;
  for (int i = tid; i < 256; i += 128) part += sw[i];
  part += __shfl_xor(part, 1, 32);
  part += __shfl_xor(part, 2, 32);
  part += __shfl_xor(part, 4, 32);
  part += __shfl_xor(part, 8, 32);
  part += __shfl_xor(part, 16, 32);
  if (lane == 0) rbuf[wave] = part;
  __syncthreads();
  if (tid == 0) atomicAdd(&den[b], rbuf[0] + rbuf[1] + rbuf[2] + rbuf[3]);

  // num[b,d] partials; column-wise re-read is L2-resident (chunk just streamed)
  float acc0 = 0.0f, acc1 = 0.0f;
  const float* colp = keyb + (size_t)s0 * D_DIM + tid;
  #pragma unroll 4
  for (int s = 0; s < 256; ++s) {
    const float wv = sw[s];
    acc0 += wv * colp[0];
    acc1 += wv * colp[128];
    colp += D_DIM;
  }
  atomicAdd(&num[(size_t)b * D_DIM + tid], acc0);
  atomicAdd(&num[(size_t)b * D_DIM + tid + 128], acc1);
}

// ---------------------------------------------------------------------------
// Finalize: out = num / (den + eps)
// ---------------------------------------------------------------------------
__global__ void finalize_kernel(const float* __restrict__ num,
                                const float* __restrict__ den,
                                float* __restrict__ out) {
  const int idx = blockIdx.x * blockDim.x + threadIdx.x;
  if (idx < B_DIM * D_DIM) {
    const int b = idx >> 8;   // D=256
    out[idx] = num[idx] / (den[b] + EPSF);
  }
}

extern "C" void kernel_launch(void* const* d_in, const int* in_sizes, int n_in,
                              void* d_out, int out_size, void* d_ws, size_t ws_size,
                              hipStream_t stream) {
  (void)in_sizes; (void)n_in; (void)out_size; (void)ws_size;
  const float* key   = (const float*)d_in[0];   // (128,1024,256)
  const float* query = (const float*)d_in[1];   // (128,1)
  const int*   mask  = (const int*)d_in[2];     // (128,1024)
  const float* W     = (const float*)d_in[3];   // (256,128)
  const float* bias  = (const float*)d_in[4];   // (128,)
  const float* Wu    = (const float*)d_in[5];   // (128,128)
  float* out = (float*)d_out;                   // (128,256)

  float* wsf  = (float*)d_ws;
  float* num  = wsf;                 // 32768 floats
  float* den  = wsf + 32768;         // 128 floats
  float* uvec = wsf + 32896;         // 128 floats
  unsigned short* Wt = (unsigned short*)(wsf + 33024);  // 32768 bf16

  prep_kernel<<<64, 256, 0, stream>>>(query, W, Wu, wsf, uvec, Wt);
  attn_main_kernel<<<512, 128, 0, stream>>>(key, mask, bias, uvec, Wt, num, den);
  finalize_kernel<<<128, 256, 0, stream>>>(num, den, out);
}
